// MultiHeadAttention_84241488544280
// MI455X (gfx1250) — matrix operand order repaired and checked by reading, not compiled, not run on hardware
//
#include <hip/hip_runtime.h>

// ---------------- problem constants ----------------
#define B_N   4
#define S_LEN 2048
#define E_DIM 1024
#define H_N   16
#define D_HD  64
#define M_TOT (B_N * S_LEN)   // 8192

// ---------------- WMMA types ----------------
typedef __attribute__((ext_vector_type(16))) __bf16 v16bf;
typedef __attribute__((ext_vector_type(8)))  float  v8f;

union FragBF {
  v16bf v;
  unsigned int u[8];
};

static __device__ __forceinline__ unsigned short f2bf(float f) {
  unsigned int u = __float_as_uint(f);
  unsigned int r = u + 0x7FFFu + ((u >> 16) & 1u);   // round-to-nearest-even
  return (unsigned short)(r >> 16);
}

// async global -> LDS copy of 16 bytes per lane (ASYNCcnt tracked)
static __device__ __forceinline__ void async_copy16(unsigned lds_off,
                                                    const void* gaddr) {
  unsigned long long ga = (unsigned long long)gaddr;
  asm volatile("global_load_async_to_lds_b128 %0, %1, off"
               :
               : "v"(lds_off), "v"(ga)
               : "memory");
}
static __device__ __forceinline__ void wait_async0() {
  asm volatile("s_wait_asynccnt 0x0" ::: "memory");
}

// ---------------- cast helper (x and all weights) ----------------
__global__ __launch_bounds__(256) void cast_f32_bf16_kernel(
    const float* __restrict__ in, unsigned short* __restrict__ out, int n4) {
  int i = blockIdx.x * blockDim.x + threadIdx.x;
  if (i < n4) {
    float4 f = ((const float4*)in)[i];
    ushort4 o;
    o.x = f2bf(f.x); o.y = f2bf(f.y); o.z = f2bf(f.z); o.w = f2bf(f.w);
    ((ushort4*)out)[i] = o;
  }
}

// vT[b][h][d][s] = v[b*S+s][h*64+d]  (tiled, coalesced both sides)
__global__ __launch_bounds__(256) void transpose_v_kernel(
    const unsigned short* __restrict__ vb, unsigned short* __restrict__ vT) {
  __shared__ unsigned short t[32][34];
  const int tx = threadIdx.x & 31, ty = threadIdx.x >> 5;  // 32x8
  const int s0 = blockIdx.x * 32;
  const int d0 = blockIdx.y * 32;          // 0 or 32
  const int b  = blockIdx.z >> 4;
  const int h  = blockIdx.z & 15;
#pragma unroll
  for (int j = 0; j < 4; j++)
    t[ty + j * 8][tx] =
        vb[(size_t)(b * S_LEN + s0 + ty + j * 8) * E_DIM + h * D_HD + d0 + tx];
  __syncthreads();
#pragma unroll
  for (int j = 0; j < 4; j++)
    vT[(size_t)((b * H_N + h) * D_HD + d0 + ty + j * 8) * S_LEN + s0 + tx] =
        t[tx][ty + j * 8];
}

// ------- tiled WMMA GEMM: C[M,N] = A[M,K] * W[N,K]^T + bias (B given row-major [N][K]) -------
#define BM 128
#define BN 128
#define BK 64

template <int OUT_F32>
__global__ __launch_bounds__(256) void gemm_bf16_kernel(
    const unsigned short* __restrict__ A,   // [M][K] bf16
    const unsigned short* __restrict__ Bw,  // [N][K] bf16 (original W layout)
    const float* __restrict__ bias,         // [N] f32 (never null)
    void* __restrict__ Cout,                // bf16 or f32
    int M, int N, int K) {
  __shared__ __align__(16) unsigned short lA[2][BM * BK];  // [m][k]
  __shared__ __align__(16) unsigned short lB[2][BN * BK];  // [n][k]

  const int tid    = threadIdx.x;
  const int m0     = blockIdx.x * BM;
  const int n0     = blockIdx.y * BN;
  const int lane   = tid & 31;
  const int wid    = tid >> 5;
  const int wm     = (wid & 3) * 32;   // wave M offset in block tile
  const int wn     = (wid >> 2) * 64;  // wave N offset in block tile
  const int lane15 = lane & 15;
  const int hi     = lane >> 4;

  v8f acc[2][4];
#pragma unroll
  for (int i = 0; i < 2; i++)
#pragma unroll
    for (int j = 0; j < 4; j++)
#pragma unroll
      for (int e = 0; e < 8; e++) acc[i][j][e] = 0.0f;

  // async-stage one K-tile (both A and B are contiguous 16B chunk copies)
  auto issue_async = [&](int kk0, int buf) {
    unsigned la0 = (unsigned)(size_t)&lA[buf][0];   // LDS offset = flat addr[31:0]
    unsigned lb0 = (unsigned)(size_t)&lB[buf][0];
#pragma unroll
    for (int j = 0; j < 4; j++) {
      int c = tid + j * 256;                 // 0..1023, chunk of 8 bf16
      int row = c >> 3, col8 = (c & 7) * 8;  // tiles are [128][64]
      async_copy16(la0 + (unsigned)c * 16u,
                   A + (size_t)(m0 + row) * K + kk0 + col8);
      async_copy16(lb0 + (unsigned)c * 16u,
                   Bw + (size_t)(n0 + row) * K + kk0 + col8);
    }
  };

  issue_async(0, 0);
  wait_async0();
  __syncthreads();

  int cur = 0;
  for (int k0 = 0; k0 < K; k0 += BK) {
    // kick off next tile's async copies; they fill lA/lB[cur^1] during WMMAs
    if (k0 + BK < K) issue_async(k0 + BK, cur ^ 1);

#pragma unroll
    for (int kc2 = 0; kc2 < 2; kc2++) {
      const int kc = kc2 * 32;
      FragBF fa[2], fb[4];
#pragma unroll
      for (int mt = 0; mt < 2; mt++) {
        int m = wm + mt * 16 + lane15;
#pragma unroll
        for (int v = 0; v < 8; v++) {
          int kk = kc + ((v < 4) ? 0 : 16) + (hi ? 8 : 0) + (v & 3) * 2;
          fa[mt].u[v] = *(const unsigned int*)&lA[cur][m * BK + kk];
        }
      }
#pragma unroll
      for (int nt = 0; nt < 4; nt++) {
        int n = wn + nt * 16 + lane15;
#pragma unroll
        for (int v = 0; v < 8; v++) {
          int kk = kc + ((v < 4) ? 0 : 16) + (hi ? 8 : 0) + (v & 3) * 2;
          fb[nt].u[v] = *(const unsigned int*)&lB[cur][n * BK + kk];
        }
      }
#pragma unroll
      for (int mt = 0; mt < 2; mt++)
#pragma unroll
        for (int nt = 0; nt < 4; nt++)
          acc[mt][nt] = __builtin_amdgcn_wmma_f32_16x16x32_bf16(
              false, fa[mt].v, false, fb[nt].v, (short)0, acc[mt][nt], false, false);
    }

    wait_async0();      // next buffer fully in LDS (this wave's copies)
    __syncthreads();    // all waves done reading cur + see next buffer
    cur ^= 1;
  }

  // branch-free epilogue (C layout: VGPR v -> M = v + 8*hi, N = lane15)
#pragma unroll
  for (int mt = 0; mt < 2; mt++) {
#pragma unroll
    for (int nt = 0; nt < 4; nt++) {
      int nc = n0 + wn + nt * 16 + lane15;
      float bv = bias[nc];
#pragma unroll
      for (int v = 0; v < 8; v++) {
        int mc = m0 + wm + mt * 16 + v + (hi ? 8 : 0);
        float val = acc[mt][nt][v] + bv;
        if (OUT_F32)
          ((float*)Cout)[(size_t)mc * N + nc] = val;
        else
          ((unsigned short*)Cout)[(size_t)mc * N + nc] = f2bf(val);
      }
    }
  }
}

// ---------------- flash attention: one wave per 16-row q tile ----------------
__global__ __launch_bounds__(128) void flash_kernel(
    const unsigned short* __restrict__ qb,   // [B*S][E] bf16
    const unsigned short* __restrict__ kb,   // [B*S][E] bf16
    const unsigned short* __restrict__ vT,   // [B][H][64][S] bf16
    const int* __restrict__ mask,            // [B][S]
    unsigned short* __restrict__ ctxb) {     // [B*S][E] bf16
  __shared__ __align__(16) unsigned short lP[4][16 * 32];

  const int lane   = threadIdx.x & 31;
  const int wv     = threadIdx.x >> 5;
  const int lane15 = lane & 15;
  const int hi     = lane >> 4;
  const int q0     = (blockIdx.x * 4 + wv) * 16;
  const int h      = blockIdx.y;
  const int b      = blockIdx.z;

  // Q fragments (K = head dim 64 -> two 16x32 A fragments), direct global loads
  FragBF qf[2];
  {
    const unsigned short* qrow =
        qb + (size_t)(b * S_LEN + q0 + lane15) * E_DIM + h * D_HD;
#pragma unroll
    for (int t = 0; t < 2; t++)
#pragma unroll
      for (int v = 0; v < 8; v++) {
        int kd = t * 32 + ((v < 4) ? 0 : 16) + (hi ? 8 : 0) + (v & 3) * 2;
        qf[t].u[v] = *(const unsigned int*)(qrow + kd);
      }
  }

  v8f ctx[4];
#pragma unroll
  for (int nt = 0; nt < 4; nt++)
#pragma unroll
    for (int e = 0; e < 8; e++) ctx[nt][e] = 0.0f;

  float mrow[8], lrow[8];
#pragma unroll
  for (int v = 0; v < 8; v++) { mrow[v] = -1.0e30f; lrow[v] = 0.0f; }

  const unsigned short* kbase = kb + (size_t)b * S_LEN * E_DIM + h * D_HD;
  const unsigned short* vbase = vT + (size_t)(b * H_N + h) * D_HD * S_LEN;
  const int* mbase = mask + b * S_LEN;

  for (int key0 = 0; key0 < S_LEN; key0 += 32) {
    // prefetch next key block (K rows + V^T rows) into cache
    if (key0 + 32 < S_LEN) {
      __builtin_prefetch(kbase + (size_t)(key0 + 32 + lane) * E_DIM, 0, 0);
      __builtin_prefetch(vbase + (size_t)lane * S_LEN + key0 + 32, 0, 0);
      __builtin_prefetch(vbase + (size_t)(32 + lane) * S_LEN + key0 + 32, 0, 0);
    }

    // ---- scores: two 16x16 f32 fragments over 32 keys ----
    v8f sf[2];
#pragma unroll
    for (int t = 0; t < 2; t++)
#pragma unroll
      for (int e = 0; e < 8; e++) sf[t][e] = 0.0f;

#pragma unroll
    for (int t = 0; t < 2; t++) {
      const unsigned short* krow =
          kbase + (size_t)(key0 + t * 16 + lane15) * E_DIM;
#pragma unroll
      for (int c = 0; c < 2; c++) {
        FragBF fk;
#pragma unroll
        for (int v = 0; v < 8; v++) {
          int kd = c * 32 + ((v < 4) ? 0 : 16) + (hi ? 8 : 0) + (v & 3) * 2;   // B layout; pairs contiguous
          fk.u[v] = *(const unsigned int*)(krow + kd);
        }
        sf[t] = __builtin_amdgcn_wmma_f32_16x16x32_bf16(
            false, qf[c].v, false, fk.v, (short)0, sf[t], false, false);
      }
    }

    float madd[2];
#pragma unroll
    for (int t = 0; t < 2; t++)
      madd[t] = (mbase[key0 + t * 16 + lane15] == 0) ? -1.0e30f : 0.0f;

    // ---- online softmax (rows live in half-waves; xor<16 stays in half) ----
    float rscale[8];
#pragma unroll
    for (int v = 0; v < 8; v++) {
      float a0 = sf[0][v] * 0.125f + madd[0];
      float a1 = sf[1][v] * 0.125f + madd[1];
      float mx = fmaxf(a0, a1);
      mx = fmaxf(mx, __shfl_xor(mx, 1, 32));
      mx = fmaxf(mx, __shfl_xor(mx, 2, 32));
      mx = fmaxf(mx, __shfl_xor(mx, 4, 32));
      mx = fmaxf(mx, __shfl_xor(mx, 8, 32));
      float mnew = fmaxf(mrow[v], mx);
      float sc   = __expf(mrow[v] - mnew);
      float p0   = __expf(a0 - mnew);
      float p1   = __expf(a1 - mnew);
      int rl = v + hi * 8;
      lP[wv][rl * 32 + lane15]      = f2bf(p0);
      lP[wv][rl * 32 + 16 + lane15] = f2bf(p1);
      float ps = p0 + p1;
      ps += __shfl_xor(ps, 1, 32);
      ps += __shfl_xor(ps, 2, 32);
      ps += __shfl_xor(ps, 4, 32);
      ps += __shfl_xor(ps, 8, 32);
      lrow[v]   = lrow[v] * sc + ps;
      mrow[v]   = mnew;
      rscale[v] = sc;
    }
#pragma unroll
    for (int nt = 0; nt < 4; nt++)
#pragma unroll
      for (int v = 0; v < 8; v++) ctx[nt][v] *= rscale[v];

    // ---- P (bf16 16x32 A frag) from LDS, then ctx += P * Vtile ----
    FragBF pf;
#pragma unroll
    for (int v = 0; v < 8; v++) {
      int kd = ((v < 4) ? 0 : 16) + (hi ? 8 : 0) + (v & 3) * 2;
      pf.u[v] = *(const unsigned int*)&lP[wv][lane15 * 32 + kd];
    }
#pragma unroll
    for (int nt = 0; nt < 4; nt++) {
      FragBF fv;
      const unsigned short* vrow =
          vbase + (size_t)(nt * 16 + lane15) * S_LEN + key0;
#pragma unroll
      for (int v = 0; v < 8; v++) {
        int kk = ((v < 4) ? 0 : 16) + (hi ? 8 : 0) + (v & 3) * 2;
        fv.u[v] = *(const unsigned int*)(vrow + kk);
      }
      ctx[nt] = __builtin_amdgcn_wmma_f32_16x16x32_bf16(
          false, pf.v, false, fv.v, (short)0, ctx[nt], false, false);
    }
  }

  // ---- normalize and store ctx as bf16 ----
#pragma unroll
  for (int nt = 0; nt < 4; nt++) {
    int nc = nt * 16 + lane15;
#pragma unroll
    for (int v = 0; v < 8; v++) {
      float inv = 1.0f / lrow[v];
      int mc = q0 + v + hi * 8;
      ctxb[(size_t)(b * S_LEN + mc) * E_DIM + h * D_HD + nc] =
          f2bf(ctx[nt][v] * inv);
    }
  }
}

// ---------------- host orchestration ----------------
extern "C" void kernel_launch(void* const* d_in, const int* in_sizes, int n_in,
                              void* d_out, int out_size, void* d_ws, size_t ws_size,
                              hipStream_t stream) {
  (void)in_sizes; (void)n_in; (void)out_size; (void)ws_size;
  const float* x    = (const float*)d_in[0];
  const int*   mask = (const int*)d_in[1];
  const float* Wq   = (const float*)d_in[2];
  const float* bq   = (const float*)d_in[3];
  const float* Wk   = (const float*)d_in[4];
  const float* bk   = (const float*)d_in[5];
  const float* Wv   = (const float*)d_in[6];
  const float* bv   = (const float*)d_in[7];
  const float* Wo   = (const float*)d_in[8];
  const float* bo   = (const float*)d_in[9];

  char* ws = (char*)d_ws;
  size_t off = 0;
  auto carve = [&](size_t bytes) -> char* {
    char* p = ws + off;
    off += (bytes + 255) & ~(size_t)255;
    return p;
  };
  const size_t act_bytes = (size_t)M_TOT * E_DIM * sizeof(unsigned short);  // 16 MB
  const size_t w_bytes   = (size_t)E_DIM * E_DIM * sizeof(unsigned short);  //  2 MB

  unsigned short* xb   = (unsigned short*)carve(act_bytes);
  unsigned short* wqb  = (unsigned short*)carve(w_bytes);   // bf16 W, [N][K]
  unsigned short* wkb  = (unsigned short*)carve(w_bytes);
  unsigned short* wvb  = (unsigned short*)carve(w_bytes);
  unsigned short* wob  = (unsigned short*)carve(w_bytes);
  unsigned short* qb   = (unsigned short*)carve(act_bytes);
  unsigned short* kbuf = (unsigned short*)carve(act_bytes);
  unsigned short* vbuf = (unsigned short*)carve(act_bytes);
  unsigned short* vTb  = (unsigned short*)carve(act_bytes);
  unsigned short* ctxb = (unsigned short*)carve(act_bytes);

  const int nAct = M_TOT * E_DIM;      // 8,388,608
  const int nW   = E_DIM * E_DIM;      // 1,048,576

  cast_f32_bf16_kernel<<<(nAct / 4 + 255) / 256, 256, 0, stream>>>(x, xb, nAct / 4);
  cast_f32_bf16_kernel<<<(nW / 4 + 255) / 256, 256, 0, stream>>>(Wq, wqb, nW / 4);
  cast_f32_bf16_kernel<<<(nW / 4 + 255) / 256, 256, 0, stream>>>(Wk, wkb, nW / 4);
  cast_f32_bf16_kernel<<<(nW / 4 + 255) / 256, 256, 0, stream>>>(Wv, wvb, nW / 4);
  cast_f32_bf16_kernel<<<(nW / 4 + 255) / 256, 256, 0, stream>>>(Wo, wob, nW / 4);

  dim3 ggrid(M_TOT / BM, E_DIM / BN);
  gemm_bf16_kernel<0><<<ggrid, 256, 0, stream>>>(xb, wqb, bq, qb,   M_TOT, E_DIM, E_DIM);
  gemm_bf16_kernel<0><<<ggrid, 256, 0, stream>>>(xb, wkb, bk, kbuf, M_TOT, E_DIM, E_DIM);
  gemm_bf16_kernel<0><<<ggrid, 256, 0, stream>>>(xb, wvb, bv, vbuf, M_TOT, E_DIM, E_DIM);

  dim3 vgrid(S_LEN / 32, D_HD / 32, B_N * H_N);
  transpose_v_kernel<<<vgrid, 256, 0, stream>>>(vbuf, vTb);

  dim3 fgrid(S_LEN / 64, H_N, B_N);
  flash_kernel<<<fgrid, 128, 0, stream>>>(qb, kbuf, vTb, mask, ctxb);

  gemm_bf16_kernel<1><<<ggrid, 256, 0, stream>>>(ctxb, wob, bo, (float*)d_out,
                                                 M_TOT, E_DIM, E_DIM);
}
